// HubDiscriminator_55155970015929
// MI455X (gfx1250) — matrix operand order repaired and checked by reading, not compiled, not run on hardware
//
#include <hip/hip_runtime.h>
#include <hip/hip_bf16.h>

#define N_NODES 100000
#define N_EDGES 1600000
#define NUM_G   128
#define HID     64
#define NLAYERS 3
#define EPS_GN  1e-5f

typedef __attribute__((ext_vector_type(16))) _Float16 v16h;
typedef __attribute__((ext_vector_type(8)))  float    v8f;

// ---------------- utility fills ----------------
__global__ void fill_f32(float* __restrict__ p, float v, int n) {
  int t = blockIdx.x * blockDim.x + threadIdx.x;
  if (t < n) p[t] = v;
}

// ---------------- degree / dinv ----------------
__global__ void deg_edges(const int* __restrict__ ei, float* deg) {
  int e = blockIdx.x * blockDim.x + threadIdx.x;
  if (e < N_EDGES) atomicAdd(&deg[ei[N_EDGES + e]], 1.0f);  // col = targets
}

__global__ void finalize_dinv(float* d) {
  int n = blockIdx.x * blockDim.x + threadIdx.x;
  if (n < N_NODES) d[n] = rsqrtf(d[n]);   // deg >= 1 (self loop) always
}

// ---------------- per-graph node counts ----------------
__global__ void count_nodes(const int* __restrict__ batch, float* cnt) {
  int n = blockIdx.x * blockDim.x + threadIdx.x;
  if (n < N_NODES) atomicAdd(&cnt[batch[n]], 1.0f);
}

__global__ void clamp_cnt(float* cnt) {
  int g = threadIdx.x;
  if (g < NUM_G) cnt[g] = fmaxf(cnt[g], 1.0f);
}

// ---------------- input projection: h = x @ in_w + in_b ----------------
__global__ void input_proj(const float* __restrict__ x, const float* __restrict__ w,
                           const float* __restrict__ b, float* __restrict__ h) {
  int t = blockIdx.x * blockDim.x + threadIdx.x;
  if (t >= N_NODES * HID) return;
  int n = t >> 6, j = t & 63;
  float s = b[j];
#pragma unroll
  for (int k = 0; k < 7; k++) s += x[n * 7 + k] * w[k * HID + j];
  h[t] = s;
}

// ---------------- WMMA GEMM: C[N_NODES,64] = A[N_NODES,64] @ W[64,64] ----------------
// One wave per 16-row tile. A tiles: 16x32 f16 (2 K-steps). B: all 8 tiles of W in regs.
__global__ __launch_bounds__(256) void gemm64_wmma(const float* __restrict__ A,
                                                   const float* __restrict__ W,
                                                   float* __restrict__ C) {
  const int lane = threadIdx.x & 31;
  const int rb = blockIdx.x * 8 + (threadIdx.x >> 5);
  if (rb >= N_NODES / 16) return;
  const int lo = lane & 15;
  const int hi = lane >> 4;   // 0 or 1

  // B tiles: Bt[kt][jt], K-range = kt*32..+31, N-range = jt*16..+15
  // element e: K = kt*32 + hi*16 + e, N = jt*16 + lo
  v16h Bt[2][4];
#pragma unroll
  for (int kt = 0; kt < 2; ++kt)
#pragma unroll
    for (int jt = 0; jt < 4; ++jt) {
      const int n = jt * 16 + lo;
#pragma unroll
      for (int e = 0; e < 16; ++e)
        Bt[kt][jt][e] = (_Float16)W[(kt * 32 + hi * 16 + e) * HID + n];
    }

  // A tiles: element e: K = kt*32 + hi*16 + e, M = lo
  const int row = rb * 16 + lo;
  v16h At[2];
#pragma unroll
  for (int kt = 0; kt < 2; ++kt)
#pragma unroll
    for (int e = 0; e < 16; ++e)
      At[kt][e] = (_Float16)A[row * HID + kt * 32 + (hi * 16 + e)];

#pragma unroll
  for (int jt = 0; jt < 4; ++jt) {
    v8f acc = {};
    acc = __builtin_amdgcn_wmma_f32_16x16x32_f16(false, At[0], false, Bt[0][jt],
                                                 (short)0, acc, false, false);
    acc = __builtin_amdgcn_wmma_f32_16x16x32_f16(false, At[1], false, Bt[1][jt],
                                                 (short)0, acc, false, false);
    const int col = jt * 16 + lo;
#pragma unroll
    for (int v = 0; v < 8; ++v)              // D: M = v + 8*hi, N = lo
      C[(rb * 16 + hi * 8 + v) * HID + col] = acc[v];
  }
}

// ---------------- agg = conv_b (broadcast init) ----------------
__global__ void agg_init(float* __restrict__ agg, const float* __restrict__ bias) {
  int t = blockIdx.x * blockDim.x + threadIdx.x;
  if (t < N_NODES * HID) agg[t] = bias[t & 63];
}

// ---------------- scatter: agg[col] += xl[row] * enorm  (E real edges + N self loops)
__global__ void scatter_edges(const int* __restrict__ ei, const float* __restrict__ dinv,
                              const float* __restrict__ xl, float* agg) {
  long long t = (long long)blockIdx.x * blockDim.x + threadIdx.x;
  const long long total = (long long)(N_EDGES + N_NODES) * 16;
  if (t >= total) return;
  int e = (int)(t >> 4);
  int q = (int)(t & 15);
  int r, c; float norm;
  if (e < N_EDGES) {
    r = ei[e]; c = ei[N_EDGES + e];
    norm = dinv[r] * dinv[c];
  } else {
    r = c = e - N_EDGES;
    float d = dinv[r]; norm = d * d;
  }
  const float4 v = *(const float4*)(xl + r * HID + q * 4);
  float* dst = agg + c * HID + q * 4;
  atomicAdd(dst + 0, v.x * norm);
  atomicAdd(dst + 1, v.y * norm);
  atomicAdd(dst + 2, v.z * norm);
  atomicAdd(dst + 3, v.w * norm);
}

// ---------------- per-graph segment sum (float4 per thread) ----------------
__global__ void seg_sum4(const float* __restrict__ src, const int* __restrict__ batch,
                         float* dst) {
  int t = blockIdx.x * blockDim.x + threadIdx.x;
  if (t >= N_NODES * 16) return;
  int n = t >> 4, c0 = (t & 15) * 4;
  int b = batch[n];
  const float4 v = *(const float4*)(src + n * HID + c0);
  float* d = dst + b * HID + c0;
  atomicAdd(d + 0, v.x); atomicAdd(d + 1, v.y);
  atomicAdd(d + 2, v.z); atomicAdd(d + 3, v.w);
}

__global__ void div_cnt(const float* __restrict__ src, const float* __restrict__ cnt,
                        float* __restrict__ dst) {
  int t = blockIdx.x * blockDim.x + threadIdx.x;
  if (t < NUM_G * HID) dst[t] = src[t] / cnt[t >> 6];
}

// ---------------- GraphNorm center + variance accumulation ----------------
__global__ void ctr_var(const float* __restrict__ agg, const int* __restrict__ batch,
                        const float* __restrict__ mean, const float* __restrict__ alpha,
                        float* __restrict__ ctr, float* vsum) {
  int t = blockIdx.x * blockDim.x + threadIdx.x;
  if (t >= N_NODES * 16) return;
  int n = t >> 4, c0 = (t & 15) * 4;
  int b = batch[n];
  const float4 a = *(const float4*)(agg + n * HID + c0);
  float cv0 = a.x - alpha[c0 + 0] * mean[b * HID + c0 + 0];
  float cv1 = a.y - alpha[c0 + 1] * mean[b * HID + c0 + 1];
  float cv2 = a.z - alpha[c0 + 2] * mean[b * HID + c0 + 2];
  float cv3 = a.w - alpha[c0 + 3] * mean[b * HID + c0 + 3];
  *(float4*)(ctr + n * HID + c0) = make_float4(cv0, cv1, cv2, cv3);
  float* vd = vsum + b * HID + c0;
  atomicAdd(vd + 0, cv0 * cv0); atomicAdd(vd + 1, cv1 * cv1);
  atomicAdd(vd + 2, cv2 * cv2); atomicAdd(vd + 3, cv3 * cv3);
}

__global__ void finalize_rstd(float* vsum, const float* __restrict__ cnt) {
  int t = blockIdx.x * blockDim.x + threadIdx.x;
  if (t < NUM_G * HID) vsum[t] = rsqrtf(vsum[t] / cnt[t >> 6] + EPS_GN);
}

// ---------------- h += relu(gamma * ctr * rstd + beta) ----------------
__global__ void apply_norm(const float* __restrict__ ctr, const int* __restrict__ batch,
                           const float* __restrict__ rstd, const float* __restrict__ gamma,
                           const float* __restrict__ beta, float* __restrict__ h) {
  int t = blockIdx.x * blockDim.x + threadIdx.x;
  if (t >= N_NODES * 16) return;
  int n = t >> 4, c0 = (t & 15) * 4;
  int b = batch[n];
  const float4 cv = *(const float4*)(ctr + n * HID + c0);
  float4 hv = *(float4*)(h + n * HID + c0);
  hv.x += fmaxf(gamma[c0 + 0] * cv.x * rstd[b * HID + c0 + 0] + beta[c0 + 0], 0.f);
  hv.y += fmaxf(gamma[c0 + 1] * cv.y * rstd[b * HID + c0 + 1] + beta[c0 + 1], 0.f);
  hv.z += fmaxf(gamma[c0 + 2] * cv.z * rstd[b * HID + c0 + 2] + beta[c0 + 2], 0.f);
  hv.w += fmaxf(gamma[c0 + 3] * cv.w * rstd[b * HID + c0 + 3] + beta[c0 + 3], 0.f);
  *(float4*)(h + n * HID + c0) = hv;
}

// ---------------- graph classifier head (tiny: 1 block, 1 thread per graph) --------
__global__ void head_mlp(const float* __restrict__ gemb,
                         const float* __restrict__ w1, const float* __restrict__ b1,
                         const float* __restrict__ w2, const float* __restrict__ b2,
                         const float* __restrict__ w3, const float* __restrict__ b3,
                         float* __restrict__ logits, float* __restrict__ probs) {
  int g = threadIdx.x;
  if (g >= NUM_G) return;
  float hr[HID];
  for (int k = 0; k < HID; k++) hr[k] = gemb[g * HID + k];
  float z1[HID];
  for (int j = 0; j < HID; j++) {
    float s = b1[j];
    for (int k = 0; k < HID; k++) s += hr[k] * w1[k * HID + j];
    z1[j] = fmaxf(s, 0.f);
  }
  float z2[32];
  for (int j = 0; j < 32; j++) {
    float s = b2[j];
    for (int k = 0; k < HID; k++) s += z1[k] * w2[k * 32 + j];
    z2[j] = fmaxf(s, 0.f);
  }
  float l0 = b3[0], l1 = b3[1];
  for (int k = 0; k < 32; k++) { l0 += z2[k] * w3[k * 2 + 0]; l1 += z2[k] * w3[k * 2 + 1]; }
  logits[g * 2 + 0] = l0;
  logits[g * 2 + 1] = l1;
  float m = fmaxf(l0, l1);
  float e0 = __expf(l0 - m), e1 = __expf(l1 - m);
  float d = e0 + e1;
  probs[g * 2 + 0] = e0 / d;
  probs[g * 2 + 1] = e1 / d;
}

// ---------------- per-node hub score: sigmoid(relu(h@ns_w1+b1)@ns_w2+b2) ----------
__global__ void node_scores(const float* __restrict__ h,
                            const float* __restrict__ w1, const float* __restrict__ b1,
                            const float* __restrict__ w2, const float* __restrict__ b2,
                            float* __restrict__ out) {
  int i = blockIdx.x * blockDim.x + threadIdx.x;
  if (i >= N_NODES) return;
  float hr[HID];
#pragma unroll
  for (int k = 0; k < HID; k++) hr[k] = h[i * HID + k];
  float acc = b2[0];
  for (int j = 0; j < 32; j++) {
    float s = b1[j];
#pragma unroll
    for (int k = 0; k < HID; k++) s += hr[k] * w1[k * 32 + j];
    acc += fmaxf(s, 0.f) * w2[j];
  }
  out[i] = 1.0f / (1.0f + __expf(-acc));
}

extern "C" void kernel_launch(void* const* d_in, const int* in_sizes, int n_in,
                              void* d_out, int out_size, void* d_ws, size_t ws_size,
                              hipStream_t stream) {
  const float* x      = (const float*)d_in[0];
  const int*   ei     = (const int*)d_in[1];
  const int*   batch  = (const int*)d_in[2];
  const float* in_w   = (const float*)d_in[3];
  const float* in_b   = (const float*)d_in[4];
  const float* conv_w = (const float*)d_in[5];
  const float* conv_b = (const float*)d_in[6];
  const float* gamma  = (const float*)d_in[7];
  const float* beta   = (const float*)d_in[8];
  const float* alpha  = (const float*)d_in[9];
  const float* gc_w1  = (const float*)d_in[10];
  const float* gc_b1  = (const float*)d_in[11];
  const float* gc_w2  = (const float*)d_in[12];
  const float* gc_b2  = (const float*)d_in[13];
  const float* gc_w3  = (const float*)d_in[14];
  const float* gc_b3  = (const float*)d_in[15];
  const float* ns_w1  = (const float*)d_in[16];
  const float* ns_b1  = (const float*)d_in[17];
  const float* ns_w2  = (const float*)d_in[18];
  const float* ns_b2  = (const float*)d_in[19];

  // Output layout: logits[G,2] | probs[G,2] | gemb[G,64] | h[N,64] | scores[N]
  float* out    = (float*)d_out;
  float* logits = out;
  float* probs  = out + NUM_G * 2;
  float* gemb   = out + NUM_G * 4;
  float* h      = out + NUM_G * 4 + NUM_G * HID;      // computed in place in d_out
  float* scores = h + (long long)N_NODES * HID;

  // Workspace layout (floats)
  float* ws   = (float*)d_ws;
  float* dinv = ws;                                   // N
  float* cnt  = dinv + N_NODES;                       // G
  float* xl   = cnt + NUM_G;                          // N*HID (reused as ctr)
  float* agg  = xl + (long long)N_NODES * HID;        // N*HID
  float* msum = agg + (long long)N_NODES * HID;       // G*HID (mean)
  float* vsum = msum + NUM_G * HID;                   // G*HID (rstd) -- contiguous w/ msum
  float* gsum = vsum + NUM_G * HID;                   // G*HID

  const int BS = 256;
  auto gs = [](long long n) { return (int)((n + 255) / 256); };

  // degrees (with self loops) -> dinv = rsqrt(deg)
  fill_f32<<<gs(N_NODES), BS, 0, stream>>>(dinv, 1.0f, N_NODES);
  deg_edges<<<gs(N_EDGES), BS, 0, stream>>>(ei, dinv);
  finalize_dinv<<<gs(N_NODES), BS, 0, stream>>>(dinv);

  // per-graph node counts
  fill_f32<<<1, BS, 0, stream>>>(cnt, 0.0f, NUM_G);
  count_nodes<<<gs(N_NODES), BS, 0, stream>>>(batch, cnt);
  clamp_cnt<<<1, NUM_G, 0, stream>>>(cnt);

  // input projection
  input_proj<<<gs((long long)N_NODES * HID), BS, 0, stream>>>(x, in_w, in_b, h);

  for (int l = 0; l < NLAYERS; l++) {
    gemm64_wmma<<<(N_NODES / 16 + 7) / 8, BS, 0, stream>>>(h, conv_w + l * HID * HID, xl);
    agg_init<<<gs((long long)N_NODES * HID), BS, 0, stream>>>(agg, conv_b + l * HID);
    scatter_edges<<<gs((long long)(N_EDGES + N_NODES) * 16), BS, 0, stream>>>(ei, dinv, xl, agg);
    // GraphNorm
    fill_f32<<<gs(2 * NUM_G * HID), BS, 0, stream>>>(msum, 0.0f, 2 * NUM_G * HID); // msum+vsum
    seg_sum4<<<gs((long long)N_NODES * 16), BS, 0, stream>>>(agg, batch, msum);
    div_cnt<<<gs(NUM_G * HID), BS, 0, stream>>>(msum, cnt, msum);                  // mean
    ctr_var<<<gs((long long)N_NODES * 16), BS, 0, stream>>>(agg, batch, msum,
                                                            alpha + l * HID, xl, vsum);
    finalize_rstd<<<gs(NUM_G * HID), BS, 0, stream>>>(vsum, cnt);
    apply_norm<<<gs((long long)N_NODES * 16), BS, 0, stream>>>(xl, batch, vsum,
                                                               gamma + l * HID,
                                                               beta + l * HID, h);
  }

  // graph pooling
  fill_f32<<<gs(NUM_G * HID), BS, 0, stream>>>(gsum, 0.0f, NUM_G * HID);
  seg_sum4<<<gs((long long)N_NODES * 16), BS, 0, stream>>>(h, batch, gsum);
  div_cnt<<<gs(NUM_G * HID), BS, 0, stream>>>(gsum, cnt, gemb);

  // heads
  head_mlp<<<1, 128, 0, stream>>>(gemb, gc_w1, gc_b1, gc_w2, gc_b2, gc_w3, gc_b3,
                                  logits, probs);
  node_scores<<<gs(N_NODES), BS, 0, stream>>>(h, ns_w1, ns_b1, ns_w2, ns_b2, scores);
}
